// PVCBurdenHead_81896436400259
// MI455X (gfx1250) — matrix operand order repaired and checked by reading, not compiled, not run on hardware
//
#include <hip/hip_runtime.h>
#include <hip/hip_bf16.h>

// ---------------------------------------------------------------------------
// PVCBurdenHead on MI455X (gfx1250).
//   * Memory-bound: one coalesced pass over 256 MB episode_ctx (~11us @ 23.3TB/s).
//   * Pooling commuted with the linear projection: [B*E,512]x[512,128]
//     (17 GFLOP) collapses to [1536,512]x[512,128] (0.2 GFLOP).
//   * GEMMs run on v_wmma_f32_16x16x32_bf16 with operands pre-packed in bf16
//     fragment layout -> inner loop = 4x b128 loads + 1 wmma.
//   * mlp1 stages its W1 K-panels into LDS with the Tensor Data Mover
//     (tensor_load_to_lds + s_wait_tensorcnt), B-fragments via ds_load_b128.
// ---------------------------------------------------------------------------

typedef __attribute__((ext_vector_type(16))) __bf16 v16bf;
typedef __attribute__((ext_vector_type(8)))  __bf16 v8bf;
typedef __attribute__((ext_vector_type(4)))  __bf16 v4bf;
typedef __attribute__((ext_vector_type(8)))  float  v8f;
typedef __attribute__((ext_vector_type(4)))  unsigned int u32x4;
typedef __attribute__((ext_vector_type(8)))  int          i32x8;
typedef __attribute__((ext_vector_type(4)))  int          i32x4;

#define N_HOURLY 24
#define D_MODEL  512
#define D_POOL   128
#define BATCH    64
#define EPS      2048
#define D_IN     3584          // 512 + 24*128
#define KSTEPS1  (D_IN / 32)   // 112 k-steps for mlp1
#define KSTEPSP  (D_MODEL / 32)// 16  k-steps for pool gemm
#define CHUNK_KS 28            // k-steps per TDM-staged LDS panel (28 KB)

// ---------------------------------------------------------------------------
// Kernel A: per-(batch,hour) bin mean over episode_ctx -> bf16 pooled means.
// grid (24,64), block 128; thread t owns float4 column t. Fully coalesced.
// ---------------------------------------------------------------------------
__global__ void pool_reduce(const float* __restrict__ ctx,
                            const int* __restrict__ n_ep,
                            __bf16* __restrict__ pooled) {
    const int h = blockIdx.x;
    const int b = blockIdx.y;
    const int t = threadIdx.x;   // 0..127

    const int nep = n_ep[b];
    const int bs  = (nep / N_HOURLY) > 0 ? (nep / N_HOURLY) : 1;
    const int start = h * bs;
    int end = (h + 1) * bs;
    if (end > nep) end = nep;

    float a0 = 0.f, a1 = 0.f, a2 = 0.f, a3 = 0.f;
    const float4* base = (const float4*)(ctx + (long)b * EPS * D_MODEL);
    for (int j = start; j < end; ++j) {
        float4 v = base[(long)j * (D_MODEL / 4) + t];
        a0 += v.x; a1 += v.y; a2 += v.z; a3 += v.w;
    }
    const float inv = (end > start) ? (1.0f / (float)(end - start)) : 0.0f;
    v4bf r;
    r[0] = (__bf16)(a0 * inv); r[1] = (__bf16)(a1 * inv);
    r[2] = (__bf16)(a2 * inv); r[3] = (__bf16)(a3 * inv);
    ((v4bf*)(pooled + ((long)b * N_HOURLY + h) * D_MODEL))[t] = r;
}

// ---------------------------------------------------------------------------
// Pack a row-major fp32 weight [Kc*32, N] into bf16 WMMA B-fragments:
// out[((tn*Kc + kc)*32 + lane)*16 + e] = W[kc*32 + (lane>>4)*8 + e + (e>=8)*8, tn*16 + (lane&15)]
// ---------------------------------------------------------------------------
__global__ void pack_w_bf16(const float* __restrict__ W, __bf16* __restrict__ out,
                            int N, int Kc, int total) {
    int i = blockIdx.x * blockDim.x + threadIdx.x;
    if (i >= total) return;
    const int lane = i & 31;
    const int kc   = (i >> 5) % Kc;
    const int tn   = i / (Kc * 32);
    const int col   = tn * 16 + (lane & 15);
    const int kbase = kc * 32 + (lane >> 4) * 8;
    v16bf frag;
    #pragma unroll
    for (int e = 0; e < 16; ++e) frag[e] = (__bf16)W[(long)(kbase + e + (e >= 8 ? 8 : 0)) * N + col];
    *(v16bf*)(out + (long)i * 16) = frag;
}

// ---------------------------------------------------------------------------
// day_embed -> x_bf[:, 0:512]
// ---------------------------------------------------------------------------
__global__ void copy_day(const float* __restrict__ day, __bf16* __restrict__ x) {
    int i = blockIdx.x * blockDim.x + threadIdx.x;
    if (i < BATCH * D_MODEL) {
        int b = i / D_MODEL, d = i % D_MODEL;
        x[(long)b * D_IN + d] = (__bf16)day[i];
    }
}

// ---------------------------------------------------------------------------
// hourly = pooled @ W_pool + b_pool (0 for empty bins) -> x_bf[:, 512:].
// M=1536, N=128, K=512. One wave per 16x16 tile; grid (8, 96), block 32.
// Inner loop: 2x b128 (A, contiguous bf16) + 2x b128 (B, packed) + 1 wmma.
// ---------------------------------------------------------------------------
__global__ void pool_gemm_wmma(const __bf16* __restrict__ pooled,
                               const __bf16* __restrict__ Wp,   // packed
                               const float* __restrict__ b_pool,
                               const int* __restrict__ n_ep,
                               __bf16* __restrict__ x) {
    const int lane = threadIdx.x;
    const int tn   = blockIdx.x;   // 0..7
    const int tm   = blockIdx.y;   // 0..95
    const int l15  = lane & 15;
    const int half = lane >> 4;
    const int row  = tm * 16 + l15;
    const int col  = tn * 16 + l15;

    v8f acc = {};
    for (int kc = 0; kc < KSTEPSP; ++kc) {
        const long abase = (long)row * D_MODEL + kc * 32 + half * 8;
        v8bf lo = *(const v8bf*)(pooled + abase);
        v8bf hi = *(const v8bf*)(pooled + abase + 16);
        v16bf a = __builtin_shufflevector(lo, hi, 0,1,2,3,4,5,6,7,8,9,10,11,12,13,14,15);
        v16bf bfrag = *((const v16bf*)Wp + ((tn * KSTEPSP + kc) * 32 + lane));
        acc = __builtin_amdgcn_wmma_f32_16x16x32_bf16(
            false, a, false, bfrag, (short)0, acc, false, false);
    }

    const float bias = b_pool[col];
    #pragma unroll
    for (int v = 0; v < 8; ++v) {
        const int r   = tm * 16 + v + half * 8;
        const int b   = r / N_HOURLY;
        const int hr  = r % N_HOURLY;
        const int nep = n_ep[b];
        const int bs  = (nep / N_HOURLY) > 0 ? (nep / N_HOURLY) : 1;
        int end = (hr + 1) * bs; if (end > nep) end = nep;
        const float val = ((end - hr * bs) > 0) ? (acc[v] + bias) : 0.0f;
        x[(long)b * D_IN + D_MODEL + hr * D_POOL + col] = (__bf16)val;
    }
}

// ---------------------------------------------------------------------------
// h = gelu(x @ W1 + b1).  M=64, N=512, K=3584.
// grid = 32 (N tiles), block (32,4): wave tm handles M-tile tm.
// Packed-W1 K-panels (28 KB) are DMA'd into LDS by the Tensor Data Mover;
// B-fragments then come from LDS (ds_load_b128).
// ---------------------------------------------------------------------------
__global__ void mlp1_wmma(const __bf16* __restrict__ x,
                          const __bf16* __restrict__ Wp1,  // packed
                          const float* __restrict__ b1,
                          float* __restrict__ h) {
    __shared__ __align__(32) unsigned char smem[CHUNK_KS * 32 * 32]; // 28 KB

    const int lane = threadIdx.x;       // 0..31
    const int tm   = threadIdx.y;       // 0..3  (M tile)
    const int tn   = blockIdx.x;        // 0..31 (N tile)
    const int l15  = lane & 15;
    const int half = lane >> 4;
    const int row  = tm * 16 + l15;
    const int col  = tn * 16 + l15;

    v8f acc = {};
    for (int c = 0; c < KSTEPS1 / CHUNK_KS; ++c) {          // 4 panels
        if (tm == 0) {
            // Tensor DMA: copy 28672 B of packed W1 (this N-tile, panel c)
            // into LDS. 1-D tile of 7168 dwords (D# per CDNA5 ISA sec.8).
            const unsigned dwords  = CHUNK_KS * 32 * 32 / 4;            // 7168
            const unsigned lds_addr = (unsigned)(unsigned long long)(&smem[0]);
            const unsigned long long ga = (unsigned long long)
                (const void*)(Wp1 + ((long)tn * KSTEPS1 + c * CHUNK_KS) * 512);
            u32x4 g0;
            g0[0] = 1u;                                   // count=1
            g0[1] = lds_addr;                             // lds_addr
            g0[2] = (unsigned)ga;                         // global_addr lo
            g0[3] = (unsigned)((ga >> 32) & 0x01FFFFFFu) | (2u << 30); // hi|type=2
            i32x8 g1;
            g1[0] = (int)(2u << 16);                      // data_size = 4B
            g1[1] = (int)((dwords & 0xFFFFu) << 16);      // tensor_dim0 lo16
            g1[2] = (int)((dwords >> 16) | (1u << 16));   // dim0 hi | tensor_dim1=1
            g1[3] = (int)(dwords << 16);                  // tile_dim0
            g1[4] = 1;                                    // tile_dim1=1, tile_dim2=0
            g1[5] = (int)dwords;                          // tensor_dim0_stride lo
            g1[6] = 0;
            g1[7] = 0;
            i32x4 z4 = {0, 0, 0, 0};
            i32x8 z8 = {0, 0, 0, 0, 0, 0, 0, 0};
            __builtin_amdgcn_tensor_load_to_lds(g0, g1, z4, z4, z8, 0);
            __builtin_amdgcn_s_wait_tensorcnt(0);
        }
        __syncthreads();

        for (int ksl = 0; ksl < CHUNK_KS; ++ksl) {
            const int k0 = (c * CHUNK_KS + ksl) * 32;
            const long abase = (long)row * D_IN + k0 + half * 8;
            v8bf lo = *(const v8bf*)(x + abase);
            v8bf hi = *(const v8bf*)(x + abase + 16);
            v16bf a = __builtin_shufflevector(lo, hi, 0,1,2,3,4,5,6,7,8,9,10,11,12,13,14,15);
            v16bf bfrag = *(const v16bf*)(smem + ((ksl * 32 + lane) << 5));
            acc = __builtin_amdgcn_wmma_f32_16x16x32_bf16(
                false, a, false, bfrag, (short)0, acc, false, false);
        }
        __syncthreads();
    }

    const float bias = b1[col];
    #pragma unroll
    for (int v = 0; v < 8; ++v) {
        const int m = tm * 16 + v + half * 8;
        const float z = acc[v] + bias;
        const float g = 0.5f * z * (1.0f + erff(z * 0.70710678118654752f));
        h[(long)m * D_MODEL + col] = g;
    }
}

// ---------------------------------------------------------------------------
// out = h @ W2 + b2.  [64,512] x [512,2] — 128 dots; VALU is the right tool.
// ---------------------------------------------------------------------------
__global__ void mlp2_kernel(const float* __restrict__ h,
                            const float* __restrict__ W2,
                            const float* __restrict__ b2,
                            float* __restrict__ out) {
    const int t = threadIdx.x;     // 0..127
    const int m = t >> 1;
    const int n = t & 1;
    float acc = 0.f;
    const float* hp = h + (long)m * D_MODEL;
    #pragma unroll 8
    for (int k = 0; k < D_MODEL; ++k) acc += hp[k] * W2[k * 2 + n];
    out[m * 2 + n] = acc + b2[n];
}

// ---------------------------------------------------------------------------
extern "C" void kernel_launch(void* const* d_in, const int* in_sizes, int n_in,
                              void* d_out, int out_size, void* d_ws, size_t ws_size,
                              hipStream_t stream) {
    const float* day_embed   = (const float*)d_in[0];
    const float* episode_ctx = (const float*)d_in[1];
    const int*   n_episodes  = (const int*)  d_in[2];
    const float* W_pool      = (const float*)d_in[3];
    const float* b_pool      = (const float*)d_in[4];
    const float* W1          = (const float*)d_in[5];
    const float* b1          = (const float*)d_in[6];
    const float* W2          = (const float*)d_in[7];
    const float* b2          = (const float*)d_in[8];
    float* out = (float*)d_out;

    // Workspace layout (32B-aligned blocks):
    //   pooled_bf [1536*512] bf16 | x_bf [64*3584] bf16 | h [64*512] f32
    //   Wp_packed [512*128] bf16  | W1_packed [3584*512] bf16
    char* ws = (char*)d_ws;
    __bf16* pooled_bf = (__bf16*)ws;                        ws += (long)1536 * 512 * 2;
    __bf16* x_bf      = (__bf16*)ws;                        ws += (long)BATCH * D_IN * 2;
    float*  hbuf      = (float*) ws;                        ws += (long)BATCH * D_MODEL * 4;
    __bf16* Wp_packed = (__bf16*)ws;                        ws += (long)D_MODEL * D_POOL * 2;
    __bf16* W1_packed = (__bf16*)ws;

    // Weight packing (one-time per call; tiny)
    {
        int totP = (D_POOL / 16) * KSTEPSP * 32;            // 4096
        pack_w_bf16<<<dim3((totP + 255) / 256), dim3(256), 0, stream>>>(
            W_pool, Wp_packed, D_POOL, KSTEPSP, totP);
        int tot1 = (D_MODEL / 16) * KSTEPS1 * 32;           // 114688
        pack_w_bf16<<<dim3((tot1 + 255) / 256), dim3(256), 0, stream>>>(
            W1, W1_packed, D_MODEL, KSTEPS1, tot1);
    }

    // A: bin means (the memory-bound pass; everything else is noise)
    pool_reduce<<<dim3(N_HOURLY, BATCH), dim3(128), 0, stream>>>(
        episode_ctx, n_episodes, pooled_bf);

    // B: day_embed -> x[:, :512]
    copy_day<<<dim3((BATCH * D_MODEL + 255) / 256), dim3(256), 0, stream>>>(
        day_embed, x_bf);

    // C: hourly -> x[:, 512:]
    pool_gemm_wmma<<<dim3(D_POOL / 16, (BATCH * N_HOURLY) / 16), dim3(32), 0, stream>>>(
        pooled_bf, Wp_packed, b_pool, n_episodes, x_bf);

    // D: h = gelu(x @ W1 + b1)  (TDM-staged B panels)
    mlp1_wmma<<<dim3(D_MODEL / 16), dim3(32, 4), 0, stream>>>(
        x_bf, W1_packed, b1, hbuf);

    // E: out = h @ W2 + b2
    mlp2_kernel<<<dim3(1), dim3(128), 0, stream>>>(hbuf, W2, b2, out);
}
